// TopkRouter_2929167696381
// MI455X (gfx1250) — matrix operand order repaired and checked by reading, not compiled, not run on hardware
//
#include <hip/hip_runtime.h>
#include <hip/hip_bf16.h>
#include <cmath>

typedef __attribute__((ext_vector_type(16))) _Float16 v16h;
typedef __attribute__((ext_vector_type(4)))  _Float16 v4h;
typedef __attribute__((ext_vector_type(8)))  float    v8f;

#define D_DIM 2048
#define E_DIM 64
#define TOK_TILE 16
#define THREADS 128

// ---------------------------------------------------------------------------
// Pre-pass: convert W_topk / W_noise (f32 [E,D]) to f16 in workspace, once.
// ---------------------------------------------------------------------------
__global__ __launch_bounds__(256)
void convert_w_kernel(const float* __restrict__ Wt, const float* __restrict__ Wn,
                      _Float16* __restrict__ Wt_h, _Float16* __restrict__ Wn_h) {
    const int i = (blockIdx.x * 256 + threadIdx.x) * 4;   // E*D = 131072 elems
    float4 a = *(const float4*)(Wt + i);
    float4 b = *(const float4*)(Wn + i);
    v4h ah, bh;
    ah[0] = (_Float16)a.x; ah[1] = (_Float16)a.y; ah[2] = (_Float16)a.z; ah[3] = (_Float16)a.w;
    bh[0] = (_Float16)b.x; bh[1] = (_Float16)b.y; bh[2] = (_Float16)b.z; bh[3] = (_Float16)b.w;
    *(v4h*)(Wt_h + i) = ah;
    *(v4h*)(Wn_h + i) = bh;
}

// ---------------------------------------------------------------------------
// Fragment loaders
// ---------------------------------------------------------------------------
// A fragment (16x32 f16, MxK): lane L<16 -> row M=L,    K = k0+0..15
//                              lane L>=16 -> row M=L-16, K = k0+16..31
__device__ __forceinline__ v16h load_frag_A(const float* __restrict__ row, int k0, int lane) {
    const int base = (lane < 16) ? 0 : 16;
    float4 f0 = *(const float4*)(row + k0 + base);
    float4 f1 = *(const float4*)(row + k0 + base + 4);
    float4 f2 = *(const float4*)(row + k0 + base + 8);
    float4 f3 = *(const float4*)(row + k0 + base + 12);
    v16h a;
    a[0]  = (_Float16)f0.x; a[1]  = (_Float16)f0.y; a[2]  = (_Float16)f0.z; a[3]  = (_Float16)f0.w;
    a[4]  = (_Float16)f1.x; a[5]  = (_Float16)f1.y; a[6]  = (_Float16)f1.z; a[7]  = (_Float16)f1.w;
    a[8]  = (_Float16)f2.x; a[9]  = (_Float16)f2.y; a[10] = (_Float16)f2.z; a[11] = (_Float16)f2.w;
    a[12] = (_Float16)f3.x; a[13] = (_Float16)f3.y; a[14] = (_Float16)f3.z; a[15] = (_Float16)f3.w;
    return a;
}

// B fragment (32x16 f16, KxN): lane L<16 -> col N=L, K=k0+0..15 ; lane>=16 -> K=k0+16..31
// f16 weight copy is [E,D] row-major -> 16 contiguous halves = 32B aligned direct load.
__device__ __forceinline__ v16h load_frag_B_h(const _Float16* __restrict__ row, int k0, int lane) {
    const int klo = (lane < 16) ? 0 : 16;
    return *(const v16h*)(row + k0 + klo);
}

// Fallback: B fragment from f32 weights (used only if ws_size is too small)
__device__ __forceinline__ v16h load_frag_B_f(const float* __restrict__ row, int k0, int lane) {
    const int klo = (lane < 16) ? 0 : 16;
    float4 f0 = *(const float4*)(row + k0 + klo);
    float4 f1 = *(const float4*)(row + k0 + klo + 4);
    float4 f2 = *(const float4*)(row + k0 + klo + 8);
    float4 f3 = *(const float4*)(row + k0 + klo + 12);
    v16h b;
    b[0]  = (_Float16)f0.x; b[1]  = (_Float16)f0.y; b[2]  = (_Float16)f0.z; b[3]  = (_Float16)f0.w;
    b[4]  = (_Float16)f1.x; b[5]  = (_Float16)f1.y; b[6]  = (_Float16)f1.z; b[7]  = (_Float16)f1.w;
    b[8]  = (_Float16)f2.x; b[9]  = (_Float16)f2.y; b[10] = (_Float16)f2.z; b[11] = (_Float16)f2.w;
    b[12] = (_Float16)f3.x; b[13] = (_Float16)f3.y; b[14] = (_Float16)f3.z; b[15] = (_Float16)f3.w;
    return b;
}

// ---------------------------------------------------------------------------
// Shared epilogue: softplus*noise + logits -> LDS -> top-2 -> 2-way softmax
// ---------------------------------------------------------------------------
struct Smem {
    float comb[TOK_TILE * E_DIM];
    float sP1[TOK_TILE], sP2[TOK_TILE];
    int   sI1[TOK_TILE], sI2[TOK_TILE];
};

__device__ __forceinline__ void epilogue(v8f acc_t, v8f acc_n,
                                         const float* __restrict__ noise,
                                         float* __restrict__ probs,
                                         int* __restrict__ idx_out,
                                         int tok_base, int lane, int wave, int tid,
                                         Smem& sm) {
    // C/D layout: VGPR i holds (M = i + 8*(lane>=16), N = lane&15), expert = N + 16*wave
    const int n_exp = (lane & 15) + wave * 16;
    const int mbase = (lane < 16) ? 0 : 8;
    #pragma unroll
    for (int i = 0; i < 8; ++i) {
        const int m = i + mbase;
        const int t = tok_base + m;
        float lg = acc_t[i];
        float nz = acc_n[i];
        float nv = noise[(size_t)t * E_DIM + n_exp];
        // softplus(beta=1, thr=20); nz<=20 => 1+exp(nz) exact enough in f32
        float sp = (nz > 20.0f) ? nz : __logf(1.0f + __expf(nz));
        sm.comb[m * E_DIM + n_exp] = sp * nv + lg;
    }
    __syncthreads();

    // Top-2 per token; strict '>' keeps lowest index on ties (matches lax.top_k)
    if (tid < TOK_TILE) {
        float v1 = -INFINITY, v2 = -INFINITY;
        int   i1 = 0,         i2 = 0;
        const float* crow = sm.comb + tid * E_DIM;
        #pragma unroll 4
        for (int e = 0; e < E_DIM; ++e) {
            float v = crow[e];
            if (v > v1)      { v2 = v1; i2 = i1; v1 = v; i1 = e; }
            else if (v > v2) { v2 = v;  i2 = e; }
        }
        float e2  = __expf(v2 - v1);
        float inv = 1.0f / (1.0f + e2);
        sm.sP1[tid] = inv; sm.sP2[tid] = e2 * inv;
        sm.sI1[tid] = i1;  sm.sI2[tid] = i2;
        const int t = tok_base + tid;
        idx_out[t * 2 + 0] = i1;
        idx_out[t * 2 + 1] = i2;
    }
    __syncthreads();

    // 128 threads scatter the sparse probability tile (16x64)
    for (int j = tid; j < TOK_TILE * E_DIM; j += THREADS) {
        const int m = j >> 6;
        const int e = j & 63;
        float p = (e == sm.sI1[m]) ? sm.sP1[m] : ((e == sm.sI2[m]) ? sm.sP2[m] : 0.0f);
        probs[(size_t)(tok_base + m) * E_DIM + e] = p;
    }
}

// ---------------------------------------------------------------------------
// Main kernel: f16 weights from workspace (fast path)
// ---------------------------------------------------------------------------
__global__ __launch_bounds__(THREADS)
void topk_router_kernel(const float* __restrict__ x,
                        const _Float16* __restrict__ Wt_h,
                        const _Float16* __restrict__ Wn_h,
                        const float* __restrict__ noise,
                        float* __restrict__ probs,
                        int* __restrict__ idx_out) {
    __shared__ Smem sm;
    const int tid      = threadIdx.x;
    const int lane     = tid & 31;
    const int wave     = tid >> 5;
    const int tok_base = blockIdx.x * TOK_TILE;
    const int mrow     = lane & 15;

    const float*    xrow   = x    + (size_t)(tok_base + mrow) * D_DIM;
    const _Float16* wt_row = Wt_h + (size_t)(wave * 16 + mrow) * D_DIM;
    const _Float16* wn_row = Wn_h + (size_t)(wave * 16 + mrow) * D_DIM;

    v8f acc_t = {};
    v8f acc_n = {};

    #pragma unroll 2
    for (int k0 = 0; k0 < D_DIM; k0 += 32) {
        __builtin_prefetch(xrow + k0 + 256, 0, 1);   // global_prefetch_b8: next x chunk
        v16h a  = load_frag_A(xrow, k0, lane);
        v16h bt = load_frag_B_h(wt_row, k0, lane);
        v16h bn = load_frag_B_h(wn_row, k0, lane);
        acc_t = __builtin_amdgcn_wmma_f32_16x16x32_f16(
            false, a, false, bt, (short)0, acc_t, false, false);
        acc_n = __builtin_amdgcn_wmma_f32_16x16x32_f16(
            false, a, false, bn, (short)0, acc_n, false, false);
    }
    epilogue(acc_t, acc_n, noise, probs, idx_out, tok_base, lane, wave, tid, sm);
}

// Fallback kernel: f32 weights with inline conversion (no workspace needed)
__global__ __launch_bounds__(THREADS)
void topk_router_kernel_f32(const float* __restrict__ x,
                            const float* __restrict__ Wt,
                            const float* __restrict__ Wn,
                            const float* __restrict__ noise,
                            float* __restrict__ probs,
                            int* __restrict__ idx_out) {
    __shared__ Smem sm;
    const int tid      = threadIdx.x;
    const int lane     = tid & 31;
    const int wave     = tid >> 5;
    const int tok_base = blockIdx.x * TOK_TILE;
    const int mrow     = lane & 15;

    const float* xrow   = x  + (size_t)(tok_base + mrow) * D_DIM;
    const float* wt_row = Wt + (size_t)(wave * 16 + mrow) * D_DIM;
    const float* wn_row = Wn + (size_t)(wave * 16 + mrow) * D_DIM;

    v8f acc_t = {};
    v8f acc_n = {};

    for (int k0 = 0; k0 < D_DIM; k0 += 32) {
        v16h a  = load_frag_A(xrow, k0, lane);
        v16h bt = load_frag_B_f(wt_row, k0, lane);
        v16h bn = load_frag_B_f(wn_row, k0, lane);
        acc_t = __builtin_amdgcn_wmma_f32_16x16x32_f16(
            false, a, false, bt, (short)0, acc_t, false, false);
        acc_n = __builtin_amdgcn_wmma_f32_16x16x32_f16(
            false, a, false, bn, (short)0, acc_n, false, false);
    }
    epilogue(acc_t, acc_n, noise, probs, idx_out, tok_base, lane, wave, tid, sm);
}

// ---------------------------------------------------------------------------
extern "C" void kernel_launch(void* const* d_in, const int* in_sizes, int n_in,
                              void* d_out, int out_size, void* d_ws, size_t ws_size,
                              hipStream_t stream) {
    const float* x       = (const float*)d_in[0];   // [B,S,D] f32
    const float* W_topk  = (const float*)d_in[1];   // [E,D]   f32
    const float* W_noise = (const float*)d_in[2];   // [E,D]   f32
    const float* noise   = (const float*)d_in[3];   // [B,S,E] f32

    const int n_tokens = in_sizes[0] / D_DIM;       // B*S = 16384

    float* probs  = (float*)d_out;                            // [B,S,E] f32
    int*   idxout = (int*)(probs + (size_t)n_tokens * E_DIM); // [B,S,2] i32

    dim3 grid(n_tokens / TOK_TILE);
    dim3 block(THREADS);

    const size_t w_elems  = (size_t)E_DIM * D_DIM;            // 131072 per matrix
    const size_t ws_need  = 2 * w_elems * sizeof(_Float16);   // 512 KB

    if (ws_size >= ws_need) {
        _Float16* Wt_h = (_Float16*)d_ws;
        _Float16* Wn_h = Wt_h + w_elems;
        // one-shot weight conversion (131072 elems / 4 per thread / 256 per block)
        convert_w_kernel<<<dim3(w_elems / (256 * 4)), dim3(256), 0, stream>>>(
            W_topk, W_noise, Wt_h, Wn_h);
        topk_router_kernel<<<grid, block, 0, stream>>>(
            x, Wt_h, Wn_h, noise, probs, idxout);
    } else {
        topk_router_kernel_f32<<<grid, block, 0, stream>>>(
            x, W_topk, W_noise, noise, probs, idxout);
    }
}
